// PointCloudGenerator_16999480557938
// MI455X (gfx1250) — matrix operand order repaired and checked by reading, not compiled, not run on hardware
//
#include <hip/hip_runtime.h>
#include <math.h>

typedef __attribute__((ext_vector_type(16))) _Float16 v16h;
typedef __attribute__((ext_vector_type(8)))  float    v8f;
typedef __attribute__((ext_vector_type(4)))  float    f32x4;

#define G        4096
#define NPTS     16384
#define NBATCH   16
#define CIN      64
#define K1PAD    96     // 66 padded up to 3 x 32
#define HDIM     256
#define NT       64     // points per tile
#define LDI      104    // sInp stride in halves (96 + 8 pad -> conflict-free)
#define LDH      264    // sH stride in halves (256 + 8 pad)
#define REG_R    0.108253175473054830f  // sqrt(3)/16

// ---------------------------------------------------------------------------
// 1) per-batch density -> base counts + fractional parts + remainder
// ---------------------------------------------------------------------------
__global__ __launch_bounds__(256) void k_prep(const float* __restrict__ dens,
                                              float* __restrict__ frac,
                                              int* __restrict__ baseCnt,
                                              int* __restrict__ rem) {
  __shared__ float red[256];
  __shared__ int   ired[256];
  const int b = blockIdx.x, t = threadIdx.x;
  const float* d = dens + (size_t)b * G;
  float s = 0.f;
  for (int c = t; c < G; c += 256) s += d[c];
  red[t] = s; __syncthreads();
  for (int o = 128; o; o >>= 1) { if (t < o) red[t] += red[t + o]; __syncthreads(); }
  const float sum = red[0];
  __syncthreads();
  int ib = 0;
  for (int c = t; c < G; c += 256) {
    float sc = d[c] / sum * (float)NPTS;
    float fb = floorf(sc);
    frac[(size_t)b * G + c] = sc - fb;
    int bi = (int)fb;
    baseCnt[(size_t)b * G + c] = bi;
    ib += bi;
  }
  ired[t] = ib; __syncthreads();
  for (int o = 128; o; o >>= 1) { if (t < o) ired[t] += ired[t + o]; __syncthreads(); }
  if (t == 0) rem[b] = NPTS - ired[0];
}

// ---------------------------------------------------------------------------
// 2) largest-remainder rank (stable ties by index) -> final counts in-place
// ---------------------------------------------------------------------------
__global__ __launch_bounds__(256) void k_rank(const float* __restrict__ frac,
                                              const int* __restrict__ rem,
                                              int* __restrict__ counts) {
  __shared__ float sF[G];
  const int b = blockIdx.x;
  const int c = blockIdx.y * 256 + threadIdx.x;
  const float* fr = frac + (size_t)b * G;
  for (int j = threadIdx.x; j < G; j += 256) sF[j] = fr[j];
  __syncthreads();
  const float f = sF[c];
  int cnt = 0;
  for (int j = 0; j < G; ++j) {
    float fj = sF[j];
    cnt += (fj > f) || (fj == f && j < c);   // rank by descending frac, stable
  }
  const int add = (cnt < rem[b]) ? 1 : 0;
  counts[(size_t)b * G + c] += add;
}

// ---------------------------------------------------------------------------
// 3) exclusive prefix over 4096 counts -> starts (CSR row pointer)
// ---------------------------------------------------------------------------
__global__ __launch_bounds__(256) void k_scan(const int* __restrict__ counts,
                                              int* __restrict__ starts) {
  __shared__ int part[256];
  const int b = blockIdx.x, t = threadIdx.x;
  const int* cb = counts + (size_t)b * G;
  int loc[16];
  int s = 0;
  for (int i = 0; i < 16; ++i) { loc[i] = s; s += cb[t * 16 + i]; }
  part[t] = s; __syncthreads();
  for (int o = 1; o < 256; o <<= 1) {
    int v = (t >= o) ? part[t - o] : 0;
    __syncthreads();
    part[t] += v;
    __syncthreads();
  }
  const int excl = part[t] - s;
  for (int i = 0; i < 16; ++i) starts[(size_t)b * G + t * 16 + i] = excl + loc[i];
}

// ---------------------------------------------------------------------------
// 4) convert weights fp32 -> fp16 (W1 zero-padded 66 -> 96)
// ---------------------------------------------------------------------------
__global__ __launch_bounds__(256) void k_wconv(const float* __restrict__ W1,
                                               const float* __restrict__ W2,
                                               unsigned short* __restrict__ W1h,
                                               unsigned short* __restrict__ W2h) {
  const int r = blockIdx.x, t = threadIdx.x;
  _Float16* w1 = (_Float16*)W1h;
  _Float16* w2 = (_Float16*)W2h;
  if (t < K1PAD) {
    float v = (t < 66) ? W1[r * 66 + t] : 0.f;
    w1[r * K1PAD + t] = (_Float16)v;
  }
  w2[r * HDIM + t] = (_Float16)W2[(size_t)r * HDIM + t];
}

// ---------------------------------------------------------------------------
// helpers for WMMA fragment loads (layouts per CDNA5 ISA 7.12.2)
// ---------------------------------------------------------------------------
__device__ __forceinline__ v16h load_fragA(const _Float16* __restrict__ W, int ldk,
                                           int m0, int k0, int llo, int lhi) {
  // A 16x32 f16: lanes 0-15 row m0+llo hold K {k0..k0+7, k0+16..k0+23};
  // lanes 16-31 same rows hold K {k0+8..k0+15, k0+24..k0+31}.
  const _Float16* p = W + (size_t)(m0 + llo) * ldk + k0 + lhi * 8;
  union { v16h v; f32x4 f[2]; } u;
  u.f[0] = *(const f32x4*)(p);
  u.f[1] = *(const f32x4*)(p + 16);
  return u.v;
}

__device__ __forceinline__ v16h load_fragB_lds(const _Float16* p32) {
  // B 32x16 f16: lanes 0-15 column N hold K {k0..k0+7, k0+16..k0+23};
  // lanes 16-31 hold K {k0+8..k0+15, k0+24..k0+31} (caller offsets pointer by lhi*8).
  union { v16h v; f32x4 f[2]; } u;
  u.f[0] = *(const f32x4*)(p32);
  u.f[1] = *(const f32x4*)(p32 + 16);
  return u.v;
}

__device__ __forceinline__ v8f bias_init(const float* __restrict__ bvec, int m) {
  // WMMA C/D layout: VGPR g holds row M=m+g (uniform over N) -> fold bias into C.
  f32x4 b0 = *(const f32x4*)(bvec + m);
  f32x4 b1 = *(const f32x4*)(bvec + m + 4);
  v8f a;
  a[0] = b0[0]; a[1] = b0[1]; a[2] = b0[2]; a[3] = b0[3];
  a[4] = b1[0]; a[5] = b1[1]; a[6] = b1[2]; a[7] = b1[3];
  return a;
}

// ---------------------------------------------------------------------------
// 5) fused: cell lookup -> gather -> L1 WMMA -> L2 WMMA -> L3 -> out/reg
//    grid (256 tiles, 16 batches) x 256 threads (8 waves)
// ---------------------------------------------------------------------------
__global__ __launch_bounds__(256, 1) void k_gemm(const float* __restrict__ x,
                                                 const float* __restrict__ noise,
                                                 const float* __restrict__ b1,
                                                 const float* __restrict__ b2,
                                                 const float* __restrict__ W3,
                                                 const float* __restrict__ b3,
                                                 const int* __restrict__ starts,
                                                 const unsigned short* __restrict__ W1h,
                                                 const unsigned short* __restrict__ W2h,
                                                 float* __restrict__ out) {
  __shared__ __align__(16) _Float16 sInp[NT][LDI];
  __shared__ __align__(16) _Float16 sH[NT][LDH];
  __shared__ float sP[4][NT][4];    // layer-3 K-split partials (padded)
  __shared__ int sCell[NT];

  const int b    = blockIdx.y;
  const int n0g  = blockIdx.x * NT;
  const int t    = threadIdx.x;
  const int wave = t >> 5;
  const int lane = t & 31;
  const int lhi  = lane >> 4;
  const int llo  = lane & 15;

  // --- per-point cell via binary search of CSR starts ---
  if (t < NT) {
    const int n = n0g + t;
    const int* st = starts + (size_t)b * G;
    int lo = 0, hi = G - 1;
    while (lo < hi) {
      int mid = (lo + hi + 1) >> 1;
      if (st[mid] <= n) lo = mid; else hi = mid - 1;
    }
    sCell[t] = lo;
  }
  __syncthreads();

  // --- gather gen_inp tile [NT x 96] into LDS (f16) ---
  {
    const int n = t >> 2, sub = t & 3;
    const int cell = sCell[n];
    const float* xb = x + (size_t)b * CIN * G;
    for (int k = sub; k < K1PAD; k += 4) {
      float v;
      if (k < CIN)      v = xb[(size_t)k * G + cell];
      else if (k < 66)  v = noise[((size_t)b * 2 + (k - CIN)) * NPTS + n0g + n] * 2.f - 1.f;
      else              v = 0.f;
      sInp[n][k] = (_Float16)v;
    }
  }
  __syncthreads();

  v8f acc[2][4];

  // ================= layer 1: [256 x 96] @ [96 x 64] =================
  {
    v8f c0 = bias_init(b1, wave * 32 + lhi * 8);
    v8f c1 = bias_init(b1, wave * 32 + 16 + lhi * 8);
    for (int j = 0; j < 4; ++j) { acc[0][j] = c0; acc[1][j] = c1; }
    const _Float16* W1p = (const _Float16*)W1h;
    for (int k0 = 0; k0 < K1PAD; k0 += 32) {
      v16h a0 = load_fragA(W1p, K1PAD, wave * 32,      k0, llo, lhi);
      v16h a1 = load_fragA(W1p, K1PAD, wave * 32 + 16, k0, llo, lhi);
      for (int j = 0; j < 4; ++j) {
        v16h bf = load_fragB_lds(&sInp[j * 16 + llo][k0 + lhi * 8]);
        acc[0][j] = __builtin_amdgcn_wmma_f32_16x16x32_f16(false, a0, false, bf,
                                                           (short)0, acc[0][j], false, false);
        acc[1][j] = __builtin_amdgcn_wmma_f32_16x16x32_f16(false, a1, false, bf,
                                                           (short)0, acc[1][j], false, false);
      }
    }
  }
  // relu -> sH (f16); each lane holds 8 consecutive M for one point
  for (int i = 0; i < 2; ++i) {
    const int m = wave * 32 + i * 16 + lhi * 8;
    for (int j = 0; j < 4; ++j) {
      const int n = j * 16 + llo;
      union { _Float16 h[8]; f32x4 f[1]; } u;
      #pragma unroll
      for (int g = 0; g < 8; ++g)
        u.h[g] = (_Float16)fmaxf(acc[i][j][g], 0.f);
      *(f32x4*)&sH[n][m] = u.f[0];
    }
  }
  __syncthreads();

  // ================= layer 2: [256 x 256] @ [256 x 64] =================
  {
    v8f c0 = bias_init(b2, wave * 32 + lhi * 8);
    v8f c1 = bias_init(b2, wave * 32 + 16 + lhi * 8);
    for (int j = 0; j < 4; ++j) { acc[0][j] = c0; acc[1][j] = c1; }
    const _Float16* W2p = (const _Float16*)W2h;
    for (int k0 = 0; k0 < HDIM; k0 += 32) {
      v16h a0 = load_fragA(W2p, HDIM, wave * 32,      k0, llo, lhi);
      v16h a1 = load_fragA(W2p, HDIM, wave * 32 + 16, k0, llo, lhi);
      for (int j = 0; j < 4; ++j) {
        v16h bf = load_fragB_lds(&sH[j * 16 + llo][k0 + lhi * 8]);
        acc[0][j] = __builtin_amdgcn_wmma_f32_16x16x32_f16(false, a0, false, bf,
                                                           (short)0, acc[0][j], false, false);
        acc[1][j] = __builtin_amdgcn_wmma_f32_16x16x32_f16(false, a1, false, bf,
                                                           (short)0, acc[1][j], false, false);
      }
    }
  }
  __syncthreads();   // all reads of h1 done before overwriting with h2
  for (int i = 0; i < 2; ++i) {
    const int m = wave * 32 + i * 16 + lhi * 8;
    for (int j = 0; j < 4; ++j) {
      const int n = j * 16 + llo;
      union { _Float16 h[8]; f32x4 f[1]; } u;
      #pragma unroll
      for (int g = 0; g < 8; ++g)
        u.h[g] = (_Float16)fmaxf(acc[i][j][g], 0.f);
      *(f32x4*)&sH[n][m] = u.f[0];
    }
  }
  __syncthreads();

  // ============ layer 3 (K split 4 ways over all 256 threads) ============
  {
    const int n  = t & (NT - 1);
    const int sl = t >> 6;           // 0..3, K slice of 64
    const int kb = sl * 64;
    float s0 = 0.f, s1 = 0.f, s2 = 0.f;
    #pragma unroll 8
    for (int k = kb; k < kb + 64; ++k) {
      const float hv = (float)sH[n][k];
      s0 += W3[k] * hv;
      s1 += W3[HDIM + k] * hv;
      s2 += W3[2 * HDIM + k] * hv;
    }
    sP[sl][n][0] = s0; sP[sl][n][1] = s1; sP[sl][n][2] = s2;
  }
  __syncthreads();

  if (t < NT) {
    const int n = t;
    const int cell = sCell[n];
    float s0 = b3[0] + sP[0][n][0] + sP[1][n][0] + sP[2][n][0] + sP[3][n][0];
    float s1 = b3[1] + sP[0][n][1] + sP[1][n][1] + sP[2][n][1] + sP[3][n][1];
    float s2 = b3[2] + sP[0][n][2] + sP[1][n][2] + sP[2][n][2] + sP[3][n][2];
    const int ci = cell >> 8, cj = (cell >> 4) & 15, ck = cell & 15;
    const float ox = ((float)ci + 0.5f) / 16.f - 0.5f;
    const float oy = ((float)cj + 0.5f) / 16.f - 0.5f;
    const float oz = ((float)ck + 0.5f) / 16.f - 0.5f;
    const float nrm = sqrtf(s0 * s0 + s1 * s1 + s2 * s2);
    const float rg  = fmaxf(nrm - REG_R, 0.f);
    const size_t base = ((size_t)b * 3) * NPTS + n0g + n;
    out[base]            = s0 + ox;
    out[base + NPTS]     = s1 + oy;
    out[base + 2 * NPTS] = s2 + oz;
    out[(size_t)NBATCH * 3 * NPTS + (size_t)b * NPTS + n0g + n] = rg;
  }
}

// ---------------------------------------------------------------------------
// host launcher
// ---------------------------------------------------------------------------
extern "C" void kernel_launch(void* const* d_in, const int* in_sizes, int n_in,
                              void* d_out, int out_size, void* d_ws, size_t ws_size,
                              hipStream_t stream) {
  (void)in_sizes; (void)n_in; (void)out_size; (void)ws_size;
  const float* x     = (const float*)d_in[0];
  const float* dens  = (const float*)d_in[1];
  const float* noise = (const float*)d_in[2];
  const float* W1    = (const float*)d_in[3];
  const float* b1    = (const float*)d_in[4];
  const float* W2    = (const float*)d_in[5];
  const float* b2    = (const float*)d_in[6];
  const float* W3    = (const float*)d_in[7];
  const float* b3    = (const float*)d_in[8];
  float* out = (float*)d_out;

  char* ws = (char*)d_ws;
  const size_t OFF_FRAC   = 0;                               // 16*4096*4
  const size_t OFF_BASE   = OFF_FRAC + (size_t)NBATCH * G * 4;
  const size_t OFF_REM    = OFF_BASE + (size_t)NBATCH * G * 4;
  const size_t OFF_STARTS = OFF_REM + 256;
  const size_t OFF_W1H    = OFF_STARTS + (size_t)NBATCH * G * 4;
  const size_t OFF_W2H    = OFF_W1H + (size_t)HDIM * K1PAD * 2;

  float*          wsFrac   = (float*)(ws + OFF_FRAC);
  int*            wsBase   = (int*)(ws + OFF_BASE);
  int*            wsRem    = (int*)(ws + OFF_REM);
  int*            wsStarts = (int*)(ws + OFF_STARTS);
  unsigned short* wsW1h    = (unsigned short*)(ws + OFF_W1H);
  unsigned short* wsW2h    = (unsigned short*)(ws + OFF_W2H);

  k_prep <<<NBATCH, 256, 0, stream>>>(dens, wsFrac, wsBase, wsRem);
  k_rank <<<dim3(NBATCH, G / 256), 256, 0, stream>>>(wsFrac, wsRem, wsBase);
  k_scan <<<NBATCH, 256, 0, stream>>>(wsBase, wsStarts);
  k_wconv<<<HDIM, 256, 0, stream>>>(W1, W2, wsW1h, wsW2h);
  k_gemm <<<dim3(NPTS / NT, NBATCH), 256, 0, stream>>>(x, noise, b1, b2, W3, b3,
                                                       wsStarts, wsW1h, wsW2h, out);
}
